// GraphTransformerEncoder_34943853920639
// MI455X (gfx1250) — matrix operand order repaired and checked by reading, not compiled, not run on hardware
//
#include <hip/hip_runtime.h>
#include <hip/hip_bf16.h>

#define NHEADS 8
#define FH 128
#define NLAYERS 6

typedef __attribute__((ext_vector_type(16))) __bf16 v16bf;
typedef __attribute__((ext_vector_type(8)))  float  v8f;

// ---------------------------------------------------------------------------
// Fused q/k/v/skip GEMM: out = x @ W + b, using v_wmma_f32_16x16x32_bf16.
// 256-thread block (8 waves). The block stages W (K x 128 f32) into LDS as
// bf16, pre-swizzled into WMMA B-fragment order so each fragment is one
// contiguous 32-byte LDS read. Each wave computes a 16(M) x 128(N) strip.
// blockIdx.y selects which of the four weight/bias/output triples.
// ---------------------------------------------------------------------------
__global__ __launch_bounds__(256) void gt_gemm_qkvs_wmma(
    const float* __restrict__ x, int K, int Nn,
    const float* __restrict__ Wq, const float* __restrict__ bq, float* __restrict__ oq,
    const float* __restrict__ Wk, const float* __restrict__ bk, float* __restrict__ ok,
    const float* __restrict__ Wv, const float* __restrict__ bv, float* __restrict__ ov,
    const float* __restrict__ Ws, const float* __restrict__ bs, float* __restrict__ os)
{
    // LDS: B fragments, [ktile(<=4)][coltile(8)][lane(32)][elem(16)] bf16 = 32KB
    __shared__ __bf16 Bsh[4 * 8 * 32 * 16];

    const float* W; const float* bias; float* out;
    switch (blockIdx.y) {
        case 0:  W = Wq; bias = bq; out = oq; break;
        case 1:  W = Wk; bias = bk; out = ok; break;
        case 2:  W = Wv; bias = bv; out = ov; break;
        default: W = Ws; bias = bs; out = os; break;
    }

    // ---- Stage + swizzle W into LDS (coalesced global reads) ----
    const int tot = K * FH;
    for (int idx = threadIdx.x; idx < tot; idx += 256) {
        const int k    = idx >> 7;          // row of W
        const int col  = idx & 127;         // col of W
        const int kt   = k >> 5;            // 32-wide K tile
        const int rem  = k & 31;
        const int half = (rem >> 3) & 1;    // B layout: half h, i<8 -> K 8h+i, i>=8 -> K 16+8h+(i-8)
        const int i    = (rem & 7) + ((rem >> 4) << 3);
        const int lane = half * 16 + (col & 15);
        const int t    = col >> 4;
        Bsh[((((kt << 3) + t) << 5) + lane) * 16 + i] = (__bf16)W[idx];
    }
    __syncthreads();

    const int lane = threadIdx.x & 31;
    const int wave = threadIdx.x >> 5;
    const int r0   = blockIdx.x * 128 + wave * 16;
    if (r0 >= Nn) return;                   // wave-uniform exit (after barrier)

    const int half = lane >> 4;
    const int l16  = lane & 15;

    v8f zero = {};
    v8f acc[8];
#pragma unroll
    for (int t = 0; t < 8; ++t) acc[t] = zero;

    int row = r0 + l16;
    if (row >= Nn) row = Nn - 1;            // clamp loads; stores are guarded
    const float4* xrow = (const float4*)(x + (size_t)row * K);

    for (int kt4 = 0; kt4 < (K >> 5); ++kt4) {
        const int kb = kt4 << 5;
        // A fragment (16x32 bf16): i=0..7 -> K kb+half*8.., i=8..15 -> K kb+16+half*8..
        const float4 c0 = xrow[(kb + half * 8) >> 2];
        const float4 c1 = xrow[(kb + half * 8 + 4) >> 2];
        const float4 c2 = xrow[(kb + 16 + half * 8) >> 2];
        const float4 c3 = xrow[(kb + 16 + half * 8 + 4) >> 2];
        v16bf a;
        a[0]  = (__bf16)c0.x; a[1]  = (__bf16)c0.y; a[2]  = (__bf16)c0.z; a[3]  = (__bf16)c0.w;
        a[4]  = (__bf16)c1.x; a[5]  = (__bf16)c1.y; a[6]  = (__bf16)c1.z; a[7]  = (__bf16)c1.w;
        a[8]  = (__bf16)c2.x; a[9]  = (__bf16)c2.y; a[10] = (__bf16)c2.z; a[11] = (__bf16)c2.w;
        a[12] = (__bf16)c3.x; a[13] = (__bf16)c3.y; a[14] = (__bf16)c3.z; a[15] = (__bf16)c3.w;

#pragma unroll
        for (int t = 0; t < 8; ++t) {
            const v16bf b = *(const v16bf*)&Bsh[((((kt4 << 3) + t) << 5) + lane) * 16];
            acc[t] = __builtin_amdgcn_wmma_f32_16x16x32_bf16(
                false, a, false, b, (short)0, acc[t], false, false);
        }
    }

    // D layout: lane col = l16, VGPR r -> row half*8 + r
#pragma unroll
    for (int t = 0; t < 8; ++t) {
        const int col = t * 16 + l16;
        const float bb = bias[col];
#pragma unroll
        for (int r = 0; r < 8; ++r) {
            const int m = half * 8 + r;
            if (r0 + m < Nn)
                out[(size_t)(r0 + m) * FH + col] = acc[t][r] + bb;
        }
    }
}

// ---------------------------------------------------------------------------
__global__ void gt_fill(float* __restrict__ p, float v, long long n) {
    long long i = (long long)blockIdx.x * blockDim.x + threadIdx.x;
    if (i < n) p[i] = v;
}

__global__ void gt_counts(const int* __restrict__ batch, float* __restrict__ counts, int n) {
    int i = blockIdx.x * blockDim.x + threadIdx.x;
    if (i < n) atomicAdd(&counts[batch[i]], 1.0f);
}

// s[e,h] = dot(q[dst[e],h,:], k[src[e],h,:]) / sqrt(C)
__global__ void gt_edge_score(const int* __restrict__ src, const int* __restrict__ dst,
                              const float* __restrict__ q, const float* __restrict__ k,
                              float* __restrict__ s, int Eh)
{
    int idx = blockIdx.x * blockDim.x + threadIdx.x;
    if (idx >= Eh) return;
    int e = idx >> 3, h = idx & 7;
    const float4* qd = (const float4*)(q + (size_t)dst[e] * FH + h * 16);
    const float4* ks = (const float4*)(k + (size_t)src[e] * FH + h * 16);
    float acc = 0.f;
#pragma unroll
    for (int c = 0; c < 4; ++c) {
        float4 a = qd[c], b = ks[c];
        acc += a.x * b.x + a.y * b.y + a.z * b.z + a.w * b.w;
    }
    s[idx] = acc * 0.25f;   // 1/sqrt(16)
}

__device__ inline void gt_atomic_max_f(float* addr, float val) {
    unsigned* ua = (unsigned*)addr;
    unsigned old = *ua;
    while (__uint_as_float(old) < val) {
        unsigned assumed = old;
        old = atomicCAS(ua, assumed, __float_as_uint(val));
        if (old == assumed) break;
    }
}

__global__ void gt_seg_max(const int* __restrict__ dst, const float* __restrict__ s,
                           float* __restrict__ m, int Eh)
{
    int idx = blockIdx.x * blockDim.x + threadIdx.x;
    if (idx >= Eh) return;
    int e = idx >> 3, h = idx & 7;
    gt_atomic_max_f(&m[(size_t)dst[e] * NHEADS + h], s[idx]);
}

__global__ void gt_exp_sum(const int* __restrict__ dst, float* __restrict__ s,
                           const float* __restrict__ m, float* __restrict__ denom, int Eh)
{
    int idx = blockIdx.x * blockDim.x + threadIdx.x;
    if (idx >= Eh) return;
    int e = idx >> 3, h = idx & 7;
    int d = dst[e];
    float ev = expf(s[idx] - m[(size_t)d * NHEADS + h]);
    s[idx] = ev;
    atomicAdd(&denom[(size_t)d * NHEADS + h], ev);
}

// msg[dst, c] += (e / max(denom,1e-16)) * v[src, c]
__global__ void gt_message(const int* __restrict__ src, const int* __restrict__ dst,
                           const float* __restrict__ s, const float* __restrict__ denom,
                           const float* __restrict__ v, float* __restrict__ msg, long long total)
{
    long long idx = (long long)blockIdx.x * blockDim.x + threadIdx.x;
    if (idx >= total) return;
    int e = (int)(idx >> 7);
    int c = (int)(idx & 127);
    int h = c >> 4;
    int d = dst[e], sc = src[e];
    float den   = denom[(size_t)d * NHEADS + h];
    float alpha = s[(size_t)e * NHEADS + h] / fmaxf(den, 1e-16f);
    atomicAdd(&msg[(size_t)d * FH + c], alpha * v[(size_t)sc * FH + c]);
}

// per-channel sum / sumsq of (msg + skip)
__global__ void gt_bn_stats(const float* __restrict__ msg, const float* __restrict__ skip,
                            float* __restrict__ stats, int n)
{
    int ch = threadIdx.x;                       // 128 threads
    int rpb = (n + gridDim.x - 1) / gridDim.x;
    int r0 = blockIdx.x * rpb;
    int r1 = min(n, r0 + rpb);
    float sum = 0.f, ssq = 0.f;
    for (int r = r0; r < r1; ++r) {
        float vl = msg[(size_t)r * FH + ch] + skip[(size_t)r * FH + ch];
        sum += vl; ssq += vl * vl;
    }
    atomicAdd(&stats[ch], sum);
    atomicAdd(&stats[FH + ch], ssq);
}

__global__ void gt_bn_finalize(const float* __restrict__ stats,
                               const float* __restrict__ gamma, const float* __restrict__ beta,
                               float* __restrict__ ss, float invN)
{
    int ch = threadIdx.x;
    float mu  = stats[ch] * invN;
    float var = stats[FH + ch] * invN - mu * mu;
    float rstd = rsqrtf(var + 1e-5f);
    float sc = gamma[ch] * rstd;
    ss[ch]      = sc;
    ss[FH + ch] = beta[ch] - mu * sc;
}

// h = relu((msg+skip)*scale + shift); pooled[g] += h
__global__ void gt_bn_apply_pool(const float* __restrict__ msg, const float* __restrict__ skip,
                                 const float* __restrict__ ss, const int* __restrict__ batch,
                                 float* __restrict__ hout, float* __restrict__ pooled, int n)
{
    long long idx = (long long)blockIdx.x * blockDim.x + threadIdx.x;
    if (idx >= (long long)n * FH) return;
    int r  = (int)(idx >> 7);
    int ch = (int)(idx & 127);
    float vl = msg[idx] + skip[idx];
    float y  = fmaxf(vl * ss[ch] + ss[FH + ch], 0.f);
    hout[idx] = y;
    atomicAdd(&pooled[(size_t)batch[r] * FH + ch], y);
}

__global__ void gt_final(const float* __restrict__ pooled, const float* __restrict__ counts,
                         float* __restrict__ out, int Gg)
{
    int idx = blockIdx.x * blockDim.x + threadIdx.x;
    if (idx >= Gg * 8 * FH) return;
    int ch = idx & 127;
    int i  = (idx >> 7) & 7;
    int g  = idx >> 10;
    const int sel[8] = {0, 1, 2, 3, 4, 5, 5, 5};
    out[idx] = pooled[((size_t)sel[i] * Gg + g) * FH + ch] / fmaxf(counts[g], 1.0f);
}

// ---------------------------------------------------------------------------
extern "C" void kernel_launch(void* const* d_in, const int* in_sizes, int n_in,
                              void* d_out, int out_size, void* d_ws, size_t ws_size,
                              hipStream_t stream)
{
    const float* x     = (const float*)d_in[0];
    const int*   ei    = (const int*)d_in[1];
    const int*   batch = (const int*)d_in[2];
    const float* gamma = (const float*)d_in[19];
    const float* beta  = (const float*)d_in[20];

    const int F0 = 64;
    const int Nn = in_sizes[0] / F0;
    const int Ee = in_sizes[1] / 2;
    const int Gg = out_size / (8 * FH);

    const int* srcI = ei;
    const int* dstI = ei + Ee;

    float* ws = (float*)d_ws;
    size_t o = 0;
    auto alloc = [&](size_t n) { float* p = ws + o; o += n; return p; };
    float* hbuf0  = alloc((size_t)Nn * FH);
    float* hbuf1  = alloc((size_t)Nn * FH);
    float* qb     = alloc((size_t)Nn * FH);
    float* kb     = alloc((size_t)Nn * FH);
    float* vb     = alloc((size_t)Nn * FH);
    float* skipb  = alloc((size_t)Nn * FH);
    float* msgb   = alloc((size_t)Nn * FH);
    float* sbuf   = alloc((size_t)Ee * NHEADS);
    float* mbuf   = alloc((size_t)Nn * NHEADS);
    float* denom  = alloc((size_t)Nn * NHEADS);
    float* stats  = alloc(2 * FH);
    float* ssb    = alloc(2 * FH);
    float* pooled = alloc((size_t)NLAYERS * Gg * FH);
    float* counts = alloc((size_t)Gg);
    (void)ws_size;

    auto cdiv = [](long long a, long long b) { return (unsigned)((a + b - 1) / b); };

    gt_fill<<<cdiv((long long)NLAYERS * Gg * FH, 256), 256, 0, stream>>>(
        pooled, 0.f, (long long)NLAYERS * Gg * FH);
    gt_fill<<<cdiv(Gg, 256), 256, 0, stream>>>(counts, 0.f, (long long)Gg);
    gt_counts<<<cdiv(Nn, 256), 256, 0, stream>>>(batch, counts, Nn);

    const float* hin = x;
    int K = F0;
    for (int layer = 0; layer < NLAYERS; ++layer) {
        const float *Wq_, *bq_, *Wk_, *bk_, *Wv_, *bv_, *Ws_, *bs_;
        if (layer == 0) {
            Wq_ = (const float*)d_in[3];  bq_ = (const float*)d_in[4];
            Wk_ = (const float*)d_in[5];  bk_ = (const float*)d_in[6];
            Wv_ = (const float*)d_in[7];  bv_ = (const float*)d_in[8];
            Ws_ = (const float*)d_in[9];  bs_ = (const float*)d_in[10];
        } else {
            size_t wo = (size_t)(layer - 1) * FH * FH;
            size_t bo = (size_t)(layer - 1) * FH;
            Wq_ = (const float*)d_in[11] + wo; bq_ = (const float*)d_in[12] + bo;
            Wk_ = (const float*)d_in[13] + wo; bk_ = (const float*)d_in[14] + bo;
            Wv_ = (const float*)d_in[15] + wo; bv_ = (const float*)d_in[16] + bo;
            Ws_ = (const float*)d_in[17] + wo; bs_ = (const float*)d_in[18] + bo;
        }

        dim3 ggrid(cdiv(Nn, 128), 4);
        gt_gemm_qkvs_wmma<<<ggrid, 256, 0, stream>>>(hin, K, Nn,
            Wq_, bq_, qb, Wk_, bk_, kb, Wv_, bv_, vb, Ws_, bs_, skipb);

        const int Eh = Ee * NHEADS;
        gt_edge_score<<<cdiv(Eh, 256), 256, 0, stream>>>(srcI, dstI, qb, kb, sbuf, Eh);

        gt_fill<<<cdiv((long long)Nn * NHEADS, 256), 256, 0, stream>>>(
            mbuf, -3.402823466e+38f, (long long)Nn * NHEADS);
        gt_fill<<<cdiv((long long)Nn * NHEADS, 256), 256, 0, stream>>>(
            denom, 0.f, (long long)Nn * NHEADS);
        gt_fill<<<cdiv((long long)Nn * FH, 256), 256, 0, stream>>>(
            msgb, 0.f, (long long)Nn * FH);
        gt_fill<<<1, 256, 0, stream>>>(stats, 0.f, (long long)(2 * FH));

        gt_seg_max<<<cdiv(Eh, 256), 256, 0, stream>>>(dstI, sbuf, mbuf, Eh);
        gt_exp_sum<<<cdiv(Eh, 256), 256, 0, stream>>>(dstI, sbuf, mbuf, denom, Eh);

        long long total = (long long)Ee * FH;
        gt_message<<<cdiv(total, 256), 256, 0, stream>>>(srcI, dstI, sbuf, denom, vb, msgb, total);

        gt_bn_stats<<<256, FH, 0, stream>>>(msgb, skipb, stats, Nn);
        gt_bn_finalize<<<1, FH, 0, stream>>>(stats, gamma + (size_t)layer * FH,
                                             beta + (size_t)layer * FH, ssb, 1.0f / (float)Nn);

        float* hout = (layer & 1) ? hbuf1 : hbuf0;
        gt_bn_apply_pool<<<cdiv((long long)Nn * FH, 256), 256, 0, stream>>>(
            msgb, skipb, ssb, batch, hout, pooled + (size_t)layer * Gg * FH, Nn);

        hin = hout;
        K = FH;
    }

    gt_final<<<cdiv((long long)Gg * 8 * FH, 256), 256, 0, stream>>>(
        pooled, counts, (float*)d_out, Gg);
}
